// EncoderLayer_80101140070420
// MI455X (gfx1250) — compile-verified, matrix operand order checked by reading, not run on hardware
//
#include <hip/hip_runtime.h>
#include <hip/hip_bf16.h>
#include <math.h>

#define DEV_INLINE __device__ __forceinline__

typedef __attribute__((ext_vector_type(16))) __bf16       bf16x16;
typedef __attribute__((ext_vector_type(8)))  float        f32x8;
typedef __attribute__((ext_vector_type(4)))  unsigned int u32x4;
typedef __attribute__((ext_vector_type(8)))  unsigned int u32x8;

union FragU { u32x4 u[2]; bf16x16 v; };
union VecU  { u32x4 u;    __bf16  h[8]; };

constexpr int kB = 512, kS = 64, kD = 768, kH = 24, kDK = 32, kDFF = 1024;

// Load a 16x32 bf16 fragment (A-operand layout, also used for transposed-B)
// from row-major storage with row stride `ld` (elements).
// Lane layout per CDNA5 ISA 7.12.2 (16-bit A, 16x32):
//   lanes 0-15 : M=lane,    K in {0..7, 16..23}
//   lanes 16-31: M=lane-16, K in {8..15, 24..31}
DEV_INLINE bf16x16 load_frag_16x32(const __bf16* src, int ld) {
  const int lane = threadIdx.x & 31;
  const int m    = lane & 15;
  const int koff = (lane & 16) ? 8 : 0;
  FragU f;
  f.u[0] = *(const u32x4*)(src + (size_t)m * ld + koff);
  f.u[1] = *(const u32x4*)(src + (size_t)m * ld + koff + 16);
  return f.v;
}

// ---------------------------------------------------------------------------
// Tensor Data Mover: DMA a rows x cols bf16 tile (row stride `stride` elems)
// from global memory into LDS at byte address `lds_byte_addr`.
// Descriptor packing per CDNA5 ISA 8.3 (group0) / 8.4 (group1):
//   g0: [1:0]=count=1, [63:32]=lds_addr, [120:64]=global_addr, [127:126]=type=2
//   g1: [17:16]=data_size(1=2B), [79:48]=tensor_dim0, [111:80]=tensor_dim1,
//       [127:112]=tile_dim0, [143:128]=tile_dim1, [207:160]=tensor_dim0_stride
// Issued per-wave (EXEC ignored); caller gates to one wave and then does
// s_wait_tensorcnt + barrier before any lane reads the tile.
// ---------------------------------------------------------------------------
DEV_INLINE void tdm_load_tile_bf16(unsigned lds_byte_addr, const __bf16* gptr,
                                   int rows, int cols, int stride) {
  unsigned long long ga = (unsigned long long)(size_t)(const void*)gptr;
  unsigned dim0 = (unsigned)stride;      // >= cols, keeps rows in-bounds
  unsigned dim1 = 0x40000000u;           // effectively unbounded in Y
  u32x4 g0;
  g0[0] = 1u;                                               // count=1
  g0[1] = lds_byte_addr;                                    // lds_addr
  g0[2] = (unsigned)ga;                                     // global_addr lo
  g0[3] = ((unsigned)(ga >> 32) & 0x01FFFFFFu) | 0x80000000u; // addr hi | type=2
  u32x8 g1;
  g1[0] = 1u << 16;                                         // data_size = 2B
  g1[1] = (dim0 & 0xFFFFu) << 16;                           // tensor_dim0 lo16
  g1[2] = (dim0 >> 16) | ((dim1 & 0xFFFFu) << 16);          // dim0 hi | dim1 lo
  g1[3] = (dim1 >> 16) | ((unsigned)cols << 16);            // dim1 hi | tile_dim0
  g1[4] = (unsigned)rows;                                   // tile_dim1 (tile_dim2=0)
  g1[5] = (unsigned)stride;                                 // tensor_dim0_stride lo32
  g1[6] = 0u;                                               // stride hi | dim1_stride lo
  g1[7] = 0u;
  asm volatile("tensor_load_to_lds %0, %1" :: "s"(g0), "s"(g1) : "memory");
}

// ---------------------------------------------------------------------------
// Generic bf16 GEMM: C[M,N] = A[M,K] @ B[K,N] (+bias) (+activation)
// Block tile 128x64, K-step 32, 256 threads = 8 waves, each wave 2x2 WMMA tiles.
// Double-buffered + software-pipelined: the TDM for A-tile i+1 is issued right
// after barrier i and runs concurrently with B-staging/WMMA of tile i.
// One barrier per K-step (safety: last reader of each buffer is one barrier
// upstream of its next writer).
// ACT: 0 = none, 1 = squared-ReLU. OutT: float or __bf16.
// Requires: M % 128 == 0, K % 32 == 0. N is guarded (handles N=32 for Wc).
// ---------------------------------------------------------------------------
template <int ACT, typename OutT>
__global__ __launch_bounds__(256) void gemm_bf16_wmma(
    const __bf16* __restrict__ A, const __bf16* __restrict__ Bm,
    const float* __restrict__ bias, OutT* __restrict__ C,
    int M, int N, int K) {
  __shared__ __bf16 As[2][128 * 32];
  __shared__ __bf16 Bs[2][64 * 32];  // stored transposed: Bs[n][k]
  const int tid  = threadIdx.x;
  const int lane = tid & 31;
  const int wave = tid >> 5;
  const int bm = blockIdx.y * 128;
  const int bn = blockIdx.x * 64;
  const int wm = (wave >> 1) * 32;  // 0,32,64,96
  const int wn = (wave & 1) * 32;   // 0,32
  const unsigned As_lds[2] = {(unsigned)(size_t)(void*)&As[0][0],
                              (unsigned)(size_t)(void*)&As[1][0]};

  f32x8 acc[2][2] = {};

  // prologue: kick off the DMA for A tile 0
  if (wave == 0)
    tdm_load_tile_bf16(As_lds[0], A + (size_t)bm * K, 128, 32, K);

  const int nk = K >> 5;
  for (int i = 0; i < nk; ++i) {
    const int ib = i & 1;
    const int k0 = i << 5;
    // --- stage B tile 32x64, transposed into Bs[ib][n][k] ---
    {
      int r  = tid >> 3;         // k row 0..31
      int c8 = (tid & 7) * 8;    // n col 0..56
      VecU t;
      const __bf16* bp = Bm + (size_t)(k0 + r) * N + (bn + c8);
      if (bn + c8 + 8 <= N) {
        t.u = *(const u32x4*)bp;
      } else {
#pragma unroll
        for (int j = 0; j < 8; ++j)
          t.h[j] = (bn + c8 + j < N) ? bp[j] : (__bf16)0.f;
      }
#pragma unroll
      for (int j = 0; j < 8; ++j) Bs[ib][(c8 + j) * 32 + r] = t.h[j];
    }
    if (k0 + 32 < K)  // prefetch next B tile (global_prefetch_b8)
      __builtin_prefetch(Bm + (size_t)(k0 + 32 + (tid >> 3)) * N + bn, 0, 0);

    __builtin_amdgcn_s_wait_tensorcnt(0);  // A tile i has landed in LDS
    __syncthreads();                       // all waves done with buffers i-1

    // overlap: DMA A tile i+1 into the other buffer while computing tile i
    if (wave == 0 && i + 1 < nk)
      tdm_load_tile_bf16(As_lds[ib ^ 1], A + (size_t)bm * K + k0 + 32,
                         128, 32, K);

    bf16x16 af[2], bfv[2];
#pragma unroll
    for (int mi = 0; mi < 2; ++mi)
      af[mi] = load_frag_16x32(&As[ib][(wm + mi * 16) * 32], 32);
#pragma unroll
    for (int ni = 0; ni < 2; ++ni)
      bfv[ni] = load_frag_16x32(&Bs[ib][(wn + ni * 16) * 32], 32);
#pragma unroll
    for (int mi = 0; mi < 2; ++mi)
#pragma unroll
      for (int ni = 0; ni < 2; ++ni)
        acc[mi][ni] = __builtin_amdgcn_wmma_f32_16x16x32_bf16(
            false, af[mi], false, bfv[ni], (short)0, acc[mi][ni], false, false);
  }

  // --- epilogue: C layout 7.12.2 (VGPR r: lanes0-15 M=r, lanes16-31 M=r+8) ---
  const int mrel = (lane & 16) ? 8 : 0;
#pragma unroll
  for (int mi = 0; mi < 2; ++mi) {
#pragma unroll
    for (int ni = 0; ni < 2; ++ni) {
      int col = bn + wn + ni * 16 + (lane & 15);
      if (col < N) {
        float bv = bias ? bias[col] : 0.f;
        int row0 = bm + wm + mi * 16 + mrel;
#pragma unroll
        for (int r = 0; r < 8; ++r) {
          float val = acc[mi][ni][r] + bv;
          if (ACT == 1) { val = fmaxf(val, 0.f); val *= val; }
          C[(size_t)(row0 + r) * N + col] = (OutT)val;
        }
      }
    }
  }
}

// ---------------------------------------------------------------------------
// Fused attention: one block (128 thr = 4 waves) per (b,h).
// Q/K tiles staged by TDM, V transposed by the waves.
// scores = (Q Kᵀ)*scale -> LDS ; softmax(scores + smol) -> P ; out = P V.
// ---------------------------------------------------------------------------
__global__ __launch_bounds__(128) void attn_wmma(
    const __bf16* __restrict__ q, const __bf16* __restrict__ k,
    const __bf16* __restrict__ v, const __bf16* __restrict__ smol,
    const float* __restrict__ qk_scale, __bf16* __restrict__ out) {
  __shared__ float  sc[64 * 64];   // scores / exp values
  __shared__ __bf16 P[64 * 64];    // softmax probs
  __shared__ __bf16 Vt[32 * 64];   // V transposed: Vt[dk][s]
  __shared__ __bf16 Qs[64 * 32];   // Q tile (row-major, ld=32)
  __shared__ __bf16 Ks[64 * 32];   // K tile (row-major, ld=32)
  const int bh   = blockIdx.x;
  const int b    = bh / kH, h = bh % kH;
  const int tid  = threadIdx.x, lane = tid & 31, wave = tid >> 5;
  const size_t base = (size_t)b * kS * kD + (size_t)h * kDK;

  // TDM: stage Q and K 64x32 tiles (row stride D)
  if (wave == 0) {
    tdm_load_tile_bf16((unsigned)(size_t)(void*)Qs, q + base, 64, 32, kD);
    tdm_load_tile_bf16((unsigned)(size_t)(void*)Ks, k + base, 64, 32, kD);
  }
  // stage V (64x32, row stride D) transposed into LDS
  {
    int s0 = tid >> 1;           // s 0..63
    int c0 = (tid & 1) * 16;     // dk 0 or 16
    VecU t0, t1;
    const __bf16* vp = v + base + (size_t)s0 * kD + c0;
    t0.u = *(const u32x4*)vp;
    t1.u = *(const u32x4*)(vp + 8);
#pragma unroll
    for (int j = 0; j < 8; ++j) {
      Vt[(c0 + j) * 64 + s0]     = t0.h[j];
      Vt[(c0 + 8 + j) * 64 + s0] = t1.h[j];
    }
  }
  __builtin_amdgcn_s_wait_tensorcnt(0);
  __syncthreads();

  const int   m0    = wave * 16;
  const int   mrel  = (lane & 16) ? 8 : 0;
  const float scale = qk_scale[0];

  // scores: wave computes rows m0..m0+15 of the 64x64 score matrix.
  bf16x16 qf = load_frag_16x32(Qs + m0 * 32, 32);
#pragma unroll
  for (int nt = 0; nt < 4; ++nt) {
    // B operand of Q@Kᵀ is K-rows (n-major), same fragment layout as A.
    bf16x16 kf = load_frag_16x32(Ks + (nt * 16) * 32, 32);
    f32x8 a = {};
    a = __builtin_amdgcn_wmma_f32_16x16x32_bf16(false, qf, false, kf,
                                                (short)0, a, false, false);
    int col = nt * 16 + (lane & 15);
#pragma unroll
    for (int r = 0; r < 8; ++r)
      sc[(m0 + mrel + r) * 64 + col] = a[r] * scale;
  }
  __syncthreads();

  // softmax(scores + smol) per row; one thread per row.
  if (tid < 64) {
    float*        srow = sc + tid * 64;
    const __bf16* brow = smol + (size_t)bh * (kS * kS) + tid * 64;
    float mx = -1e30f;
    for (int j = 0; j < 64; ++j) {
      float t = srow[j] + (float)brow[j];
      srow[j] = t;
      mx = fmaxf(mx, t);
    }
    float ssum = 0.f;
    for (int j = 0; j < 64; ++j) {
      float e = __expf(srow[j] - mx);
      srow[j] = e;
      ssum += e;
    }
    float inv = 1.f / ssum;
    __bf16* prow = P + tid * 64;
    for (int j = 0; j < 64; ++j) prow[j] = (__bf16)(srow[j] * inv);
  }
  __syncthreads();

  // out = P @ V : wave computes 16x32 (K=64 -> two k-tiles)
  f32x8 o[2] = {};
#pragma unroll
  for (int kt = 0; kt < 2; ++kt) {
    bf16x16 pf = load_frag_16x32(P + m0 * 64 + kt * 32, 64);
#pragma unroll
    for (int ni = 0; ni < 2; ++ni) {
      bf16x16 vf = load_frag_16x32(Vt + (ni * 16) * 64 + kt * 32, 64);
      o[ni] = __builtin_amdgcn_wmma_f32_16x16x32_bf16(false, pf, false, vf,
                                                      (short)0, o[ni], false, false);
    }
  }
#pragma unroll
  for (int ni = 0; ni < 2; ++ni) {
    int dk = ni * 16 + (lane & 15);
#pragma unroll
    for (int r = 0; r < 8; ++r) {
      int s = m0 + mrel + r;
      out[base + (size_t)s * kD + dk] = (__bf16)o[ni][r];
    }
  }
}

// ---------------------------------------------------------------------------
// Row LayerNorm, one block (256 thr) per row of length L (<= 6144, L%4==0).
// Row is pulled into LDS with async global->LDS copies (ASYNCcnt), then
// transformed in place.
// act==1: v = silu(A[j])           (SmolGen hidden layers)
// act==0: v = A[j] + alpha*R[j]    (residual branches; R/alpha may be null)
// ---------------------------------------------------------------------------
__global__ __launch_bounds__(256) void ln_rows(
    const float* __restrict__ A, const float* __restrict__ R,
    const float* __restrict__ alpha, const float* __restrict__ g,
    const float* __restrict__ bta, float* __restrict__ outf,
    __bf16* __restrict__ outb, int L, int act) {
  __shared__ float buf[6144];
  __shared__ float rs[256], rs2[256];
  const int    tid = threadIdx.x;
  const size_t off = (size_t)blockIdx.x * L;
  const float  al  = alpha ? alpha[0] : 0.f;

  // async copy of the row into LDS (16B per lane per issue)
  {
    const unsigned buf_lds = (unsigned)(size_t)(void*)buf;
    const int nchunk = L >> 2;
    for (int c = tid; c < nchunk; c += 256) {
      unsigned dst = buf_lds + (unsigned)c * 16u;
      unsigned long long src = (unsigned long long)(size_t)(const void*)(A + off + (size_t)c * 4);
      asm volatile("global_load_async_to_lds_b128 %0, %1, off"
                   :: "v"(dst), "v"(src) : "memory");
    }
    asm volatile("s_wait_asynccnt 0x0" ::: "memory");
    __syncthreads();
  }

  float s = 0.f, s2 = 0.f;
  for (int j = tid; j < L; j += 256) {
    float vv = buf[j];
    if (act) vv = vv / (1.f + __expf(-vv));
    else if (R) vv += al * R[off + j];
    buf[j] = vv;
    s += vv;
    s2 += vv * vv;
  }
  rs[tid] = s;
  rs2[tid] = s2;
  __syncthreads();
  for (int st = 128; st > 0; st >>= 1) {
    if (tid < st) { rs[tid] += rs[tid + st]; rs2[tid] += rs2[tid + st]; }
    __syncthreads();
  }
  const float mean = rs[0] / (float)L;
  const float var  = rs2[0] / (float)L - mean * mean;
  const float inv  = rsqrtf(var + 1e-5f);
  for (int j = tid; j < L; j += 256) {
    float y = (buf[j] - mean) * inv * g[j] + bta[j];
    if (outf) outf[off + j] = y;
    if (outb) outb[off + j] = (__bf16)y;
  }
}

__global__ __launch_bounds__(256) void cvt_f32_bf16(
    const float* __restrict__ in, __bf16* __restrict__ out, long long n) {
  long long i = (long long)blockIdx.x * blockDim.x + threadIdx.x;
  long long stride = (long long)gridDim.x * blockDim.x;
  for (; i < n; i += stride) out[i] = (__bf16)in[i];
}

// ---------------------------------------------------------------------------
extern "C" void kernel_launch(void* const* d_in, const int* in_sizes, int n_in,
                              void* d_out, int out_size, void* d_ws, size_t ws_size,
                              hipStream_t stream) {
  (void)in_sizes; (void)n_in; (void)out_size; (void)ws_size;
  const float* x    = (const float*)d_in[0];
  const float* Wq   = (const float*)d_in[1];
  const float* bq   = (const float*)d_in[2];
  const float* Wk   = (const float*)d_in[3];
  const float* bk   = (const float*)d_in[4];
  const float* Wv   = (const float*)d_in[5];
  const float* bv   = (const float*)d_in[6];
  const float* Wo   = (const float*)d_in[7];
  const float* bo   = (const float*)d_in[8];
  const float* qks  = (const float*)d_in[9];
  const float* Wc   = (const float*)d_in[10];
  const float* Wd1  = (const float*)d_in[11];
  const float* bd1  = (const float*)d_in[12];
  const float* g1   = (const float*)d_in[13];
  const float* b1   = (const float*)d_in[14];
  const float* Wd2  = (const float*)d_in[15];
  const float* bd2  = (const float*)d_in[16];
  const float* g2   = (const float*)d_in[17];
  const float* b2   = (const float*)d_in[18];
  const float* Wsg  = (const float*)d_in[19];
  const float* ln1g = (const float*)d_in[20];
  const float* ln1b = (const float*)d_in[21];
  const float* ln2g = (const float*)d_in[22];
  const float* ln2b = (const float*)d_in[23];
  const float* W1   = (const float*)d_in[24];
  const float* bf1  = (const float*)d_in[25];
  const float* W2   = (const float*)d_in[26];
  const float* bf2  = (const float*)d_in[27];
  const float* a_in = (const float*)d_in[28];
  const float* a_out= (const float*)d_in[29];
  float* outp = (float*)d_out;

  const size_t MT = (size_t)kB * kS;  // 32768 token rows

  char* w = (char*)d_ws;
  auto alloc = [&](size_t bytes) -> char* {
    char* p = w;
    w += (bytes + 255) & ~(size_t)255;
    return p;
  };
  __bf16* xb   = (__bf16*)alloc(MT * kD * 2);
  __bf16* wqb  = (__bf16*)alloc((size_t)kD * kD * 2);
  __bf16* wkb  = (__bf16*)alloc((size_t)kD * kD * 2);
  __bf16* wvb  = (__bf16*)alloc((size_t)kD * kD * 2);
  __bf16* wob  = (__bf16*)alloc((size_t)kD * kD * 2);
  __bf16* wcb  = (__bf16*)alloc((size_t)kD * 32 * 2);
  __bf16* wd1b = (__bf16*)alloc((size_t)(kS * 32) * 256 * 2);
  __bf16* wd2b = (__bf16*)alloc((size_t)256 * (kH * 256) * 2);
  __bf16* wsgb = (__bf16*)alloc((size_t)256 * (kS * kS) * 2);
  __bf16* w1b  = (__bf16*)alloc((size_t)kD * kDFF * 2);
  __bf16* w2b  = (__bf16*)alloc((size_t)kDFF * kD * 2);
  __bf16* qb   = (__bf16*)alloc(MT * kD * 2);
  __bf16* kb   = (__bf16*)alloc(MT * kD * 2);
  __bf16* vb   = (__bf16*)alloc(MT * kD * 2);
  __bf16* cb   = (__bf16*)alloc(MT * 32 * 2);               // == [B, 2048] bf16
  float*  h1f  = (float*) alloc((size_t)kB * 256 * 4);
  __bf16* hb   = (__bf16*)alloc((size_t)kB * 256 * 2);
  float*  h2f  = (float*) alloc((size_t)kB * (kH * 256) * 4);
  __bf16* h2b  = (__bf16*)alloc((size_t)kB * (kH * 256) * 2);
  __bf16* smol = (__bf16*)alloc((size_t)kB * kH * kS * kS * 2);
  __bf16* attnb= (__bf16*)alloc(MT * kD * 2);
  float*  bigf = (float*) alloc(MT * kD * 4);               // wo_out, later ff2
  float*  x1f  = (float*) alloc(MT * kD * 4);
  __bf16* x1b  = (__bf16*)alloc(MT * kD * 2);
  __bf16* ff1b = (__bf16*)alloc(MT * kDFF * 2);

  auto cvt = [&](const float* src, __bf16* dst, size_t n) {
    long long nn = (long long)n;
    int blocks = (int)((nn + 256 * 8 - 1) / (256 * 8));
    if (blocks > 4096) blocks = 4096;
    cvt_f32_bf16<<<blocks, 256, 0, stream>>>(src, dst, nn);
  };
  auto ggrid = [](int M, int N) {
    return dim3((unsigned)((N + 63) / 64), (unsigned)((M + 127) / 128));
  };

  // 1) cast activations + weights to bf16
  cvt(x,   xb,   MT * kD);
  cvt(Wq,  wqb,  (size_t)kD * kD);
  cvt(Wk,  wkb,  (size_t)kD * kD);
  cvt(Wv,  wvb,  (size_t)kD * kD);
  cvt(Wo,  wob,  (size_t)kD * kD);
  cvt(Wc,  wcb,  (size_t)kD * 32);
  cvt(Wd1, wd1b, (size_t)(kS * 32) * 256);
  cvt(Wd2, wd2b, (size_t)256 * (kH * 256));
  cvt(Wsg, wsgb, (size_t)256 * (kS * kS));
  cvt(W1,  w1b,  (size_t)kD * kDFF);
  cvt(W2,  w2b,  (size_t)kDFF * kD);

  // 2) QKV projections
  gemm_bf16_wmma<0, __bf16><<<ggrid((int)MT, kD), 256, 0, stream>>>(
      xb, wqb, bq, qb, (int)MT, kD, kD);
  gemm_bf16_wmma<0, __bf16><<<ggrid((int)MT, kD), 256, 0, stream>>>(
      xb, wkb, bk, kb, (int)MT, kD, kD);
  gemm_bf16_wmma<0, __bf16><<<ggrid((int)MT, kD), 256, 0, stream>>>(
      xb, wvb, bv, vb, (int)MT, kD, kD);

  // 3) SmolGen: c = x@Wc  ([B*S,32] contiguous == [B,2048])
  gemm_bf16_wmma<0, __bf16><<<ggrid((int)MT, 32), 256, 0, stream>>>(
      xb, wcb, nullptr, cb, (int)MT, 32, kD);
  // h = LN(silu(c@Wd1 + bd1))
  gemm_bf16_wmma<0, float><<<ggrid(kB, 256), 256, 0, stream>>>(
      cb, wd1b, bd1, h1f, kB, 256, kS * 32);
  ln_rows<<<kB, 256, 0, stream>>>(h1f, nullptr, nullptr, g1, b1,
                                  nullptr, hb, 256, 1);
  // h2 = LN(silu(h@Wd2 + bd2))  (LN over full 6144)
  gemm_bf16_wmma<0, float><<<ggrid(kB, kH * 256), 256, 0, stream>>>(
      hb, wd2b, bd2, h2f, kB, kH * 256, 256);
  ln_rows<<<kB, 256, 0, stream>>>(h2f, nullptr, nullptr, g2, b2,
                                  nullptr, h2b, kH * 256, 1);
  // smol = h2([B*H,256]) @ Wsg
  gemm_bf16_wmma<0, __bf16><<<ggrid(kB * kH, kS * kS), 256, 0, stream>>>(
      h2b, wsgb, nullptr, smol, kB * kH, kS * kS, 256);

  // 4) fused attention (softmax(QKᵀ*scale + smol) @ V)
  attn_wmma<<<kB * kH, 128, 0, stream>>>(qb, kb, vb, smol, qks, attnb);

  // 5) output projection + residual LN
  gemm_bf16_wmma<0, float><<<ggrid((int)MT, kD), 256, 0, stream>>>(
      attnb, wob, bo, bigf, (int)MT, kD, kD);
  ln_rows<<<(int)MT, 256, 0, stream>>>(bigf, x, a_in, ln1g, ln1b,
                                       x1f, x1b, kD, 0);

  // 6) squared-ReLU FFN + residual LN
  gemm_bf16_wmma<1, __bf16><<<ggrid((int)MT, kDFF), 256, 0, stream>>>(
      x1b, w1b, bf1, ff1b, (int)MT, kDFF, kD);
  gemm_bf16_wmma<0, float><<<ggrid((int)MT, kD), 256, 0, stream>>>(
      ff1b, w2b, bf2, bigf, (int)MT, kD, kDFF);
  ln_rows<<<(int)MT, 256, 0, stream>>>(bigf, x1f, a_out, ln2g, ln2b,
                                       outp, nullptr, kD, 0);
}
